// GIIF_module_88957362635281
// MI455X (gfx1250) — matrix operand order repaired and checked by reading, not compiled, not run on hardware
//
#include <hip/hip_runtime.h>

// MI455X / gfx1250, wave32. Dual-stream cross attention:
//   B=8, L=2048, C=256, D=32.  ~88 GFLOP total; everything-fp16 fits in L2.
// v_wmma_f32_16x16x32_f16 for all GEMMs; flash-style online softmax.
// Attention stages K/V tiles into LDS with the Tensor Data Mover
// (tensor_load_to_lds, TENSORcnt, double buffered) so the 8 waves of a block
// share each tile: cuts L2 traffic 8x (4.6GB -> 0.6GB) and overlaps DMA with
// the WMMA body.

typedef _Float16 v16h __attribute__((ext_vector_type(16)));
typedef _Float16 h8   __attribute__((ext_vector_type(8)));
typedef float    v8f  __attribute__((ext_vector_type(8)));
typedef unsigned int u32x4 __attribute__((ext_vector_type(4)));
typedef int          i32x4 __attribute__((ext_vector_type(4)));
typedef int          i32x8 __attribute__((ext_vector_type(8)));

#define B_ 8
#define L_ 2048
#define C_ 256
#define D_ 32

#if __has_builtin(__builtin_amdgcn_tensor_load_to_lds)
#define USE_TDM 1
#else
#define USE_TDM 0
#endif

static __device__ __forceinline__ v16h mk16(h8 lo, h8 hi) {
  v16h r;
#pragma unroll
  for (int i = 0; i < 8; ++i) { r[i] = lo[i]; r[i + 8] = hi[i]; }
  return r;
}

static __device__ __forceinline__ v8f vzero() {
  v8f r;
#pragma unroll
  for (int i = 0; i < 8; ++i) r[i] = 0.0f;
  return r;
}

static __device__ __forceinline__ v8f wmma16(v16h a, v16h b, v8f c) {
  // 8 args: (neg_a, A, neg_b, B, c_mod, C, reuse_a, reuse_b)
  return __builtin_amdgcn_wmma_f32_16x16x32_f16(false, a, false, b, (short)0, c,
                                                false, false);
}

#if USE_TDM
// TDM: DMA a 2-D tile of 2-byte elements (w_elems wide, h_rows tall, row
// stride stride_elems) from global memory into LDS at byte offset lds_off.
// D# layout per CDNA5 ISA ch.8 (group0: ctrl/lds/global/type, group1: dims).
static __device__ __forceinline__ void tdm_load_2d(const void* gptr,
                                                   unsigned lds_off,
                                                   unsigned w_elems,
                                                   unsigned h_rows,
                                                   unsigned stride_elems) {
  unsigned long long ga = (unsigned long long)(uintptr_t)gptr;
  u32x4 g0;
  g0[0] = 1u;                                  // count=1, user descriptor
  g0[1] = lds_off;                             // lds_addr (bytes)
  g0[2] = (unsigned)(ga & 0xFFFFFFFFu);        // global_addr[31:0]
  g0[3] = (unsigned)((ga >> 32) & 0x01FFFFFFu) // global_addr[56:32]
          | (2u << 30);                        // type = 2 ("image")
  i32x8 g1;
  g1[0] = (int)(1u << 16);                     // wg_mask=0, data_size=1 (2B)
  g1[1] = (int)(0xFFFFu << 16);                // tensor_dim0[15:0] (large)
  g1[2] = (int)(0xFu | (0xFFFFu << 16));       // tdim0[31:16], tdim1[15:0]
  g1[3] = (int)(0xFu | (w_elems << 16));       // tdim1[31:16], tile_dim0=w
  g1[4] = (int)(h_rows & 0xFFFFu);             // tile_dim1=h, tile_dim2=0
  g1[5] = (int)stride_elems;                   // tensor_dim0_stride[31:0]
  g1[6] = 0;                                   // stride[47:32], dim1_stride lo
  g1[7] = 0;
  i32x4 gz = {0, 0, 0, 0};
#if __clang_major__ >= 23
  i32x8 gz8 = {0, 0, 0, 0, 0, 0, 0, 0};
  __builtin_amdgcn_tensor_load_to_lds(g0, g1, gz, gz, gz8, 0);
#else
  __builtin_amdgcn_tensor_load_to_lds(g0, g1, gz, gz, 0);
#endif
}
#endif

// ---------------------------------------------------------------------------
// Kernel 1: split interleaved f32 x(B,L,C,2) into two contiguous f16 streams.
// ---------------------------------------------------------------------------
__global__ __launch_bounds__(256) void pack_x_kernel(
    const float* __restrict__ x, _Float16* __restrict__ eOut,
    _Float16* __restrict__ pOut, int n) {
  int i = blockIdx.x * 256 + threadIdx.x;
  if (i < n) {
    float2 v = ((const float2*)x)[i];
    eOut[i] = (_Float16)v.x;
    pOut[i] = (_Float16)v.y;
  }
}

// ---------------------------------------------------------------------------
// Kernel 2: pre-swizzle a weight matrix (C x dout, f32 row-major) into
// B-fragment order so each lane's v16h is 16 contiguous halves.
//   Wp[nt][ks][lane][e] = w[ks*32 + e + 16*(lane/16)][nt*16 + lane%16]
// ---------------------------------------------------------------------------
__global__ __launch_bounds__(256) void pack_w_kernel(
    const float* __restrict__ w, int dout, _Float16* __restrict__ wp,
    int nelem) {
  int i = blockIdx.x * 256 + threadIdx.x;
  if (i >= nelem) return;
  int e    = i & 15;
  int lane = (i >> 4) & 31;
  int ks   = (i >> 9) & 7;
  int nt   = i >> 12;
  int k = ks * 32 + e + ((lane >> 4) << 4);
  int n = nt * 16 + (lane & 15);
  wp[i] = (_Float16)w[(size_t)k * dout + n];
}

// ---------------------------------------------------------------------------
// Kernel 3: projection GEMM.  A:(16384 x 256) f16, W packed, K=256 (8 wmma).
// One wave computes one 16x16 output tile.
// transposed==0: out row-major (16384 x dout)   (Q/K tensors, dout=32)
// transposed==1: out channel-major Vt (B,256,L) (V tensors, dout=256)
// ---------------------------------------------------------------------------
__global__ __launch_bounds__(256) void proj_kernel(
    const _Float16* __restrict__ A, const _Float16* __restrict__ Wp,
    _Float16* __restrict__ out, int dout, int transposed) {
  const int lane = threadIdx.x & 31;
  const int wave = threadIdx.x >> 5;
  const int tile = blockIdx.x * 8 + wave;
  const int mt = tile & 1023;   // 1024 M-tiles of 16 rows
  const int nt = tile >> 10;
  const int g  = lane >> 4;
  const int ln = lane & 15;

  const _Float16* arow  = A + (size_t)(mt * 16 + ln) * C_;
  const _Float16* wbase = Wp + (size_t)nt * 4096 + lane * 16;

  v8f acc = vzero();
#pragma unroll
  for (int ks = 0; ks < 8; ++ks) {
    h8 alo = *(const h8*)(arow + ks * 32 + g * 16);
    h8 ahi = *(const h8*)(arow + ks * 32 + g * 16 + 8);
    const _Float16* wl = wbase + ks * 512;
    h8 blo = *(const h8*)(wl);
    h8 bhi = *(const h8*)(wl + 8);
    acc = wmma16(mk16(alo, ahi), mk16(blo, bhi), acc);
  }

  const int n = nt * 16 + ln;
  if (transposed) {
    int r0 = mt * 16 + g * 8;               // 8 consecutive global rows
    int b  = r0 / L_;
    int rr = r0 % L_;
    h8 t;
#pragma unroll
    for (int i = 0; i < 8; ++i) t[i] = (_Float16)acc[i];
    *(h8*)(out + (size_t)(b * C_ + n) * L_ + rr) = t;
  } else {
#pragma unroll
    for (int i = 0; i < 8; ++i) {
      int r = mt * 16 + g * 8 + i;
      out[(size_t)r * dout + n] = (_Float16)acc[i];
    }
  }
}

// ---------------------------------------------------------------------------
// Kernel 4: flash attention. 8 waves/block; wave = 16 query rows; the block
// shares TDM-staged K/V tiles (32 keys/step) from LDS, double buffered.
// Q:(16384 x 32) f16 row-major; K likewise; Vt:(B,256,L) f16 channel-major.
// intra passes: out = scale*O/l ; inter passes: out += O/l.
// ---------------------------------------------------------------------------
__global__ __launch_bounds__(256) void attn_kernel(
    const _Float16* __restrict__ Q, const _Float16* __restrict__ K,
    const _Float16* __restrict__ Vt, float* __restrict__ out, int outoff,
    const float* __restrict__ scale_ptr, int accum) {
  __shared__ __attribute__((aligned(16))) _Float16 kbuf[2][32][32];   // 2x2KB
  __shared__ __attribute__((aligned(16))) _Float16 vbuf[2][256][32];  // 2x16KB
  __shared__ __attribute__((aligned(16))) _Float16 ldsP[8][16][32];   // 8KB

  const int lane = threadIdx.x & 31;
  const int wave = threadIdx.x >> 5;
  const int g  = lane >> 4;
  const int ln = lane & 15;
  const int qt = blockIdx.x * 8 + wave;        // 0..1023 query tiles
  const int b  = (int)(blockIdx.x * 128) / L_; // batch (block stays in-batch)

  // A-fragment of Q (held for entire key loop).
  const _Float16* qr = Q + (size_t)(qt * 16 + ln) * D_;
  v16h qfrag = mk16(*(const h8*)(qr + g * 16), *(const h8*)(qr + g * 16 + 8));

  const _Float16* Kb = K + (size_t)b * L_ * D_;
  const _Float16* Vb = Vt + (size_t)b * C_ * L_;

  // Stage key-step tile (32 keys of K, 256x32 slab of Vt) into buffer nb.
  auto stage = [&](int kt, int nb) {
#if USE_TDM
    if (wave == 0) {
      tdm_load_2d(Kb + (size_t)(kt * 32) * D_,
                  (unsigned)(uintptr_t)&kbuf[nb][0][0], 32, 32, 32);
      tdm_load_2d(Vb + kt * 32,
                  (unsigned)(uintptr_t)&vbuf[nb][0][0], 32, 256, L_);
    }
#else
    const int t = (int)threadIdx.x;
    const _Float16* vsrc = Vb + (size_t)t * L_ + kt * 32;   // channel t
#pragma unroll
    for (int j = 0; j < 4; ++j)
      *(h8*)&vbuf[nb][t][j * 8] = *(const h8*)(vsrc + j * 8);
    if (t < 64) {
      int r = t >> 1, c = (t & 1) * 16;
      const _Float16* ksrc = Kb + (size_t)(kt * 32 + r) * D_ + c;
      *(h8*)&kbuf[nb][r][c]     = *(const h8*)(ksrc);
      *(h8*)&kbuf[nb][r][c + 8] = *(const h8*)(ksrc + 8);
    }
#endif
  };

  v8f O[16];
#pragma unroll
  for (int nt = 0; nt < 16; ++nt) O[nt] = vzero();
  float m[8], lsum[8];
#pragma unroll
  for (int i = 0; i < 8; ++i) { m[i] = -3.0e38f; lsum[i] = 0.0f; }

  const float sc = 0.17677669529663687f;  // 1/sqrt(32)

  stage(0, 0);
#if USE_TDM
  __builtin_amdgcn_s_wait_tensorcnt(0);
#endif
  __syncthreads();

#pragma unroll 1
  for (int kt = 0; kt < 64; ++kt) {
    const int cur = kt & 1;
    if (kt < 63) stage(kt + 1, cur ^ 1);  // overlap DMA with WMMA body

    // --- S tiles from LDS K: two WMMAs (keys 0..15 and 16..31 of step) ---
    const _Float16* kb0 = &kbuf[cur][ln][g * 16];
    const _Float16* kb1 = &kbuf[cur][ln + 16][g * 16];
    v16h bk0 = mk16(*(const h8*)kb0, *(const h8*)(kb0 + 8));
    v16h bk1 = mk16(*(const h8*)kb1, *(const h8*)(kb1 + 8));
    v8f s0 = wmma16(qfrag, bk0, vzero());
    v8f s1 = wmma16(qfrag, bk1, vzero());

    // --- online softmax in C-layout (row = vgpr + 8*g, reduce over 16 lanes)
    float mc[8];
#pragma unroll
    for (int i = 0; i < 8; ++i) {
      s0[i] *= sc; s1[i] *= sc;
      mc[i] = fmaxf(s0[i], s1[i]);
    }
#pragma unroll
    for (int off = 8; off >= 1; off >>= 1)
#pragma unroll
      for (int i = 0; i < 8; ++i)
        mc[i] = fmaxf(mc[i], __shfl_xor(mc[i], off, 16));

    float corr[8], rs[8];
#pragma unroll
    for (int i = 0; i < 8; ++i) {
      float mn = fmaxf(m[i], mc[i]);
      corr[i] = __expf(m[i] - mn);
      m[i] = mn;
      s0[i] = __expf(s0[i] - mn);
      s1[i] = __expf(s1[i] - mn);
      rs[i] = s0[i] + s1[i];
    }
#pragma unroll
    for (int off = 8; off >= 1; off >>= 1)
#pragma unroll
      for (int i = 0; i < 8; ++i) rs[i] += __shfl_xor(rs[i], off, 16);
#pragma unroll
    for (int i = 0; i < 8; ++i) lsum[i] = lsum[i] * corr[i] + rs[i];
#pragma unroll
    for (int nt = 0; nt < 16; ++nt)
#pragma unroll
      for (int i = 0; i < 8; ++i) O[nt][i] *= corr[i];

    // --- C-layout P -> A-layout P via LDS (per-wave private 1KB) ---
#pragma unroll
    for (int i = 0; i < 8; ++i) {
      int r = i + g * 8;
      ldsP[wave][r][ln]      = (_Float16)s0[i];
      ldsP[wave][r][ln + 16] = (_Float16)s1[i];
    }
    h8 plo = *(const h8*)&ldsP[wave][ln][g * 16];
    h8 phi = *(const h8*)&ldsP[wave][ln][g * 16 + 8];
    v16h pfrag = mk16(plo, phi);

    // --- O += P * V over the 16 value n-tiles (V from LDS) ---
#pragma unroll
    for (int nt = 0; nt < 16; ++nt) {
      const _Float16* vp = &vbuf[cur][nt * 16 + ln][g * 16];
      v16h vfrag = mk16(*(const h8*)vp, *(const h8*)(vp + 8));
      O[nt] = wmma16(pfrag, vfrag, O[nt]);
    }

#if USE_TDM
    __builtin_amdgcn_s_wait_tensorcnt(0);  // wave0: next tile DMA complete
#endif
    __syncthreads();                       // all waves done with cur buffer
  }

  // --- epilogue: normalize, scale, write/accumulate ---
  const float s = scale_ptr ? *scale_ptr : 1.0f;
  float inv[8];
#pragma unroll
  for (int i = 0; i < 8; ++i) inv[i] = s / lsum[i];
#pragma unroll
  for (int nt = 0; nt < 16; ++nt) {
#pragma unroll
    for (int i = 0; i < 8; ++i) {
      size_t idx = (size_t)(qt * 16 + g * 8 + i) * 512 + outoff + nt * 16 + ln;
      float v = O[nt][i] * inv[i];
      if (accum)
        out[idx] += v;
      else
        out[idx] = v;
    }
  }
}

// ---------------------------------------------------------------------------
// Host launcher
// ---------------------------------------------------------------------------
extern "C" void kernel_launch(void* const* d_in, const int* in_sizes, int n_in,
                              void* d_out, int out_size, void* d_ws,
                              size_t ws_size, hipStream_t stream) {
  (void)in_sizes; (void)n_in; (void)out_size; (void)ws_size;

  const float* x     = (const float*)d_in[0];
  const float* alpha = (const float*)d_in[13];
  const float* gamma = (const float*)d_in[14];
  float* out = (float*)d_out;

  _Float16* ws = (_Float16*)d_ws;
  const size_t nBLC = (size_t)B_ * L_ * C_;        // 4,194,304
  const size_t qkStride = (size_t)B_ * L_ * D_;    // 524,288
  _Float16* Ae = ws;                                // ecg f16 (16384x256)
  _Float16* Ap = Ae + nBLC;                         // pcg f16
  _Float16* QK = Ap + nBLC;                         // 8 x (16384x32)
  _Float16* Vt = QK + 8 * qkStride;                 // 4 x (B,256,L)
  _Float16* Wp = Vt + 4 * nBLC;                     // packed weights (~0.65MB)

  pack_x_kernel<<<(int)(nBLC / 256), 256, 0, stream>>>(x, Ae, Ap, (int)nBLC);

  const int dout[12] = {32, 32, 256, 32, 32, 256, 32, 32, 256, 32, 32, 256};
  size_t wpoff = 0;
  for (int t = 0; t < 12; ++t) {
    int ne = (dout[t] / 16) * 4096;
    pack_w_kernel<<<ne / 256, 256, 0, stream>>>((const float*)d_in[1 + t],
                                                dout[t], Wp + wpoff, ne);
    wpoff += (size_t)ne;
  }

  // sources: t0-3 & t10,11 from ecg; t4-9 from pcg
  const _Float16* src[12] = {Ae, Ae, Ae, Ae, Ap, Ap, Ap, Ap, Ap, Ap, Ae, Ae};
  _Float16* dst[12];
  dst[0] = QK + 0 * qkStride;  // Qe_intra
  dst[1] = QK + 1 * qkStride;  // Ke_intra
  dst[2] = Vt + 0 * nBLC;      // Ve_intra
  dst[3] = QK + 2 * qkStride;  // Qe_inter
  dst[4] = QK + 3 * qkStride;  // Kp_inter
  dst[5] = Vt + 1 * nBLC;      // Vp_inter
  dst[6] = QK + 4 * qkStride;  // Qp_intra
  dst[7] = QK + 5 * qkStride;  // Kp_intra
  dst[8] = Vt + 2 * nBLC;      // Vp_intra
  dst[9] = QK + 6 * qkStride;  // Qp_inter
  dst[10] = QK + 7 * qkStride; // Ke_inter
  dst[11] = Vt + 3 * nBLC;     // Ve_inter

  wpoff = 0;
  for (int t = 0; t < 12; ++t) {
    int nt = dout[t] / 16;
    proj_kernel<<<1024 * nt / 8, 256, 0, stream>>>(
        src[t], Wp + wpoff, dst[t], dout[t], (dout[t] == 256) ? 1 : 0);
    wpoff += (size_t)nt * 4096;
  }

  // ecg_out = inter + alpha*intra  (cols 0..255)
  attn_kernel<<<128, 256, 0, stream>>>(QK + 0 * qkStride, QK + 1 * qkStride,
                                       Vt + 0 * nBLC, out, 0, alpha, 0);
  attn_kernel<<<128, 256, 0, stream>>>(QK + 2 * qkStride, QK + 3 * qkStride,
                                       Vt + 1 * nBLC, out, 0, nullptr, 1);
  // pcg_out = inter + gamma*intra  (cols 256..511)
  attn_kernel<<<128, 256, 0, stream>>>(QK + 4 * qkStride, QK + 5 * qkStride,
                                       Vt + 2 * nBLC, out, 256, gamma, 0);
  attn_kernel<<<128, 256, 0, stream>>>(QK + 6 * qkStride, QK + 7 * qkStride,
                                       Vt + 3 * nBLC, out, 256, nullptr, 1);
}
